// mulGCN_45518063403266
// MI455X (gfx1250) — matrix operand order repaired and checked by reading, not compiled, not run on hardware
//
#include <hip/hip_runtime.h>

typedef __attribute__((ext_vector_type(16))) __bf16 v16bf;
typedef __attribute__((ext_vector_type(8)))  float  v8f;

#define N_NODES   50000
#define N_EDGES   800000
#define DDIM      256
#define GFEAT     200
#define GFEAT_PAD 208
#define NG        512
#define MTILES16  (N_NODES / 16)        // 3125 exact
#define MBLOCKS32 ((N_NODES + 31) / 32) // 1563, last block has 16 rows

// ---------- helpers ----------
static __device__ __forceinline__ unsigned short f2bf(float f) {
    unsigned u = __float_as_uint(f);
    unsigned r = 0x7fffu + ((u >> 16) & 1u);     // round-to-nearest-even
    return (unsigned short)((u + r) >> 16);
}
static __device__ __forceinline__ unsigned packbf(float lo, float hi) {
    return (unsigned)f2bf(lo) | ((unsigned)f2bf(hi) << 16);
}

// ---------- prep kernels ----------
// node feats f32[row][256] -> packed bf16 pairs uint[row][128]
__global__ __launch_bounds__(256) void pack_feats(const float* __restrict__ h,
                                                  unsigned* __restrict__ out, int n) {
    int i = blockIdx.x * blockDim.x + threadIdx.x;
    if (i >= n) return;
    out[i] = packbf(h[2 * i], h[2 * i + 1]);
}
// weight f32[256][256] -> fragment-major packed:
//   out[((kt*2+half)*256 + n)*8 + i] = pack(W[k][n], W[k+1][n]), k = 32*kt + 16*half + 2*i
__global__ __launch_bounds__(256) void pack_weight_frag(const float* __restrict__ W,
                                                        unsigned* __restrict__ out) {
    int idx = blockIdx.x * blockDim.x + threadIdx.x;
    if (idx >= 16 * 256 * 8) return;
    int i  = idx & 7;
    int n  = (idx >> 3) & 255;
    int kh = idx >> 11;                  // kt*2 + half, 0..15
    int k  = 32 * (kh >> 1) + 16 * (kh & 1) + 2 * i;
    out[idx] = packbf(W[k * 256 + n], W[(k + 1) * 256 + n]);
}
// readout weight f32[256][200] -> fragment-major packed over 208 padded cols
__global__ __launch_bounds__(256) void pack_ro_frag(const float* __restrict__ W,
                                                    unsigned* __restrict__ out) {
    int idx = blockIdx.x * blockDim.x + threadIdx.x;
    if (idx >= 16 * GFEAT_PAD * 8) return;
    int i    = idx & 7;
    int rest = idx >> 3;                 // kh*208 + n
    int n    = rest % GFEAT_PAD;
    int kh   = rest / GFEAT_PAD;
    int k    = 32 * (kh >> 1) + 16 * (kh & 1) + 2 * i;
    float lo = 0.f, hi = 0.f;
    if (n < GFEAT) { lo = W[k * GFEAT + n]; hi = W[(k + 1) * GFEAT + n]; }
    out[idx] = packbf(lo, hi);
}

// ---------- WMMA GEMM: [N_NODES x 256] = A(bf16) @ B(bf16), f32 accum ----------
// Each block: 32 rows; each wave: 32x64 slab (2 A frags x 4 B frags, 8 WMMA/k-step).
// MODE 0: outF = A@B
// MODE 1: outBf = bf16( relu(agg + b1) + relu(A@B + b2) )
// MODE 2: outBf = bf16( relu(A@B + b1) )
template <int MODE>
__global__ __launch_bounds__(128)
void gemm256(const unsigned* __restrict__ Apk,   // [M][128] packed bf16 pairs
             const unsigned* __restrict__ Bpk,   // fragment-major [16][256][8]
             const float* __restrict__ agg,
             const float* __restrict__ bias1,
             const float* __restrict__ bias2,
             float* __restrict__ outF,
             unsigned short* __restrict__ outBf) {
    const int wave = threadIdx.x >> 5;
    const int lane = threadIdx.x & 31;
    const int half = lane >> 4;
    const int ln   = lane & 15;
    const int mbase = blockIdx.x << 5;           // 32 rows per block
    const int nb    = wave << 6;                 // 64 output cols per wave
    const bool hasM2 = (mbase + 16) < N_NODES;   // wave-uniform
    const unsigned* arow0 = Apk + (size_t)(mbase + ln) * 128;
    const unsigned* arow1 = Apk + (size_t)(mbase + 16 + ln) * 128;

    union Frag { v16bf v; unsigned u[8]; };
    v8f acc[2][4];
#pragma unroll
    for (int m = 0; m < 2; ++m)
#pragma unroll
        for (int t = 0; t < 4; ++t)
#pragma unroll
            for (int r = 0; r < 8; ++r) acc[m][t][r] = 0.f;

    for (int kt = 0; kt < 8; ++kt) {             // K = 256 = 8 * 32
        Frag a0, a1;
#pragma unroll
        for (int i = 0; i < 4; ++i) a0.u[i]     = arow0[16 * kt + 8 * half + i];
#pragma unroll
        for (int i = 0; i < 4; ++i) a0.u[4 + i] = arow0[16 * kt + 8 * half + 4 + i];
        if (hasM2) {
#pragma unroll
            for (int i = 0; i < 4; ++i) a1.u[i]     = arow1[16 * kt + 8 * half + i];
#pragma unroll
            for (int i = 0; i < 4; ++i) a1.u[4 + i] = arow1[16 * kt + 8 * half + 4 + i];
        }
        const unsigned* bbase =
            Bpk + ((size_t)(kt * 2 + half) * 256 + nb + ln) * 8;
#pragma unroll
        for (int t = 0; t < 4; ++t) {
            Frag b;
#pragma unroll
            for (int i = 0; i < 8; ++i) b.u[i] = bbase[t * 128 + i];
            acc[0][t] = __builtin_amdgcn_wmma_f32_16x16x32_bf16(
                false, a0.v, false, b.v, (short)0, acc[0][t], false, false);
            if (hasM2)
                acc[1][t] = __builtin_amdgcn_wmma_f32_16x16x32_bf16(
                    false, a1.v, false, b.v, (short)0, acc[1][t], false, false);
        }
    }
    const int mEnd = hasM2 ? 2 : 1;
    for (int m = 0; m < mEnd; ++m) {
#pragma unroll
        for (int r = 0; r < 8; ++r) {
            const int row = mbase + 16 * m + r + 8 * half;
#pragma unroll
            for (int t = 0; t < 4; ++t) {
                const int col = nb + t * 16 + ln;
                const float v = acc[m][t][r];
                if (MODE == 0) {
                    outF[(size_t)row * 256 + col] = v;
                } else if (MODE == 1) {
                    float c = agg[(size_t)row * 256 + col] + bias1[col];
                    c = c > 0.f ? c : 0.f;
                    float rr = v + bias2[col];
                    rr = rr > 0.f ? rr : 0.f;
                    outBf[(size_t)row * 256 + col] = f2bf(c + rr);
                } else {
                    float c = v + bias1[col];
                    outBf[(size_t)row * 256 + col] = f2bf(c > 0.f ? c : 0.f);
                }
            }
        }
    }
}

// ---------- readout GEMM: y = A(bf16)@Ro + bo, atomically segment-summed ----------
__global__ __launch_bounds__(32)
void gemm_readout(const unsigned* __restrict__ Apk,  // [M][128]
                  const unsigned* __restrict__ Bpk,  // fragment-major [16][208][8]
                  const float* __restrict__ bo,      // [200]
                  const int* __restrict__ gids,      // [M]
                  float* __restrict__ gacc) {        // [512][200]
    const int lane = threadIdx.x & 31;
    const int half = lane >> 4;
    const int ln   = lane & 15;
    const int mbase = blockIdx.x << 4;
    const int nt    = blockIdx.y;                    // 0..12
    const unsigned* arow = Apk + (size_t)(mbase + ln) * 128;

    union Frag { v16bf v; unsigned u[8]; };
    v8f acc;
#pragma unroll
    for (int r = 0; r < 8; ++r) acc[r] = 0.f;

    for (int kt = 0; kt < 8; ++kt) {
        Frag a;
#pragma unroll
        for (int i = 0; i < 4; ++i) a.u[i]     = arow[16 * kt + 8 * half + i];
#pragma unroll
        for (int i = 0; i < 4; ++i) a.u[4 + i] = arow[16 * kt + 8 * half + 4 + i];
        const unsigned* bbase =
            Bpk + ((size_t)(kt * 2 + half) * GFEAT_PAD + nt * 16 + ln) * 8;
        Frag b;
#pragma unroll
        for (int i = 0; i < 8; ++i) b.u[i] = bbase[i];
        acc = __builtin_amdgcn_wmma_f32_16x16x32_bf16(
            false, a.v, false, b.v, (short)0, acc, false, false);
    }
    const int col = nt * 16 + ln;
    if (col < GFEAT) {
        const float bv = bo[col];
#pragma unroll
        for (int r = 0; r < 8; ++r) {
            const int row = mbase + r + 8 * half;
            atomicAdd(&gacc[(size_t)gids[row] * GFEAT + col], acc[r] + bv);
        }
    }
}

// ---------- edge aggregation: agg[dst] += hW[src] ----------
__global__ __launch_bounds__(256)
void edge_agg(const float4* __restrict__ hW, const int* __restrict__ src,
              const int* __restrict__ dst, float* __restrict__ agg) {
    const long long gid = (long long)blockIdx.x * blockDim.x + threadIdx.x;
    const int e = (int)(gid >> 6);
    if (e >= N_EDGES) return;
    const int c = (int)(gid & 63);
    const int s = src[e], d = dst[e];
    const float4 v = hW[(size_t)s * 64 + c];
    float* o = agg + (size_t)d * 256 + c * 4;
    atomicAdd(o + 0, v.x);
    atomicAdd(o + 1, v.y);
    atomicAdd(o + 2, v.z);
    atomicAdd(o + 3, v.w);
}

// ---------- predictor: out[g] = gacc[g,:] . Wp + bp ----------
__global__ __launch_bounds__(256)
void predict_k(const float* __restrict__ gacc, const float* __restrict__ Wp,
               const float* __restrict__ bp, float* __restrict__ out) {
    int g = blockIdx.x * blockDim.x + threadIdx.x;
    if (g >= NG) return;
    float s = bp[0];
    const float* row = gacc + (size_t)g * GFEAT;
    for (int f = 0; f < GFEAT; ++f) s += row[f] * Wp[f];
    out[g] = s;
}

extern "C" void kernel_launch(void* const* d_in, const int* in_sizes, int n_in,
                              void* d_out, int out_size, void* d_ws, size_t ws_size,
                              hipStream_t stream) {
    (void)in_sizes; (void)n_in; (void)out_size;
    const float* nf1  = (const float*)d_in[0];
    const float* nf2  = (const float*)d_in[2];
    const int*   src1 = (const int*)d_in[4];
    const int*   dst1 = (const int*)d_in[5];
    const int*   gid1 = (const int*)d_in[6];
    const int*   src2 = (const int*)d_in[7];
    const int*   dst2 = (const int*)d_in[8];
    const int*   gid2 = (const int*)d_in[9];
    const float* W1   = (const float*)d_in[10]; const float* b1   = (const float*)d_in[11];
    const float* Wr1  = (const float*)d_in[12]; const float* br1  = (const float*)d_in[13];
    const float* W2   = (const float*)d_in[14]; const float* b2   = (const float*)d_in[15];
    const float* Wr2  = (const float*)d_in[16]; const float* br2  = (const float*)d_in[17];
    const float* Ri1  = (const float*)d_in[18]; const float* rbi1 = (const float*)d_in[19];
    const float* Ro1  = (const float*)d_in[20]; const float* rbo1 = (const float*)d_in[21];
    const float* Ri2  = (const float*)d_in[22]; const float* rbi2 = (const float*)d_in[23];
    const float* Ro2  = (const float*)d_in[24]; const float* rbo2 = (const float*)d_in[25];
    const float* Wp   = (const float*)d_in[26]; const float* bp   = (const float*)d_in[27];
    float* out = (float*)d_out;

    // ---- workspace carve ----
    char* ws = (char*)d_ws;
    size_t off = 0;
    auto take = [&](size_t bytes) -> char* {
        off = (off + 255) & ~(size_t)255;
        char* p = ws + off;
        off += bytes;
        return p;
    };
    unsigned*       hA   = (unsigned*)take((size_t)N_NODES * 128 * 4);  // bf16 feats (packed)
    unsigned short* hB   = (unsigned short*)take((size_t)N_NODES * 256 * 2);
    float*          F1   = (float*)take((size_t)N_NODES * 256 * 4);     // hW
    float*          F2   = (float*)take((size_t)N_NODES * 256 * 4);     // agg
    unsigned* W1p  = (unsigned*)take(16 * 256 * 8 * 4);
    unsigned* Wr1p = (unsigned*)take(16 * 256 * 8 * 4);
    unsigned* W2p  = (unsigned*)take(16 * 256 * 8 * 4);
    unsigned* Wr2p = (unsigned*)take(16 * 256 * 8 * 4);
    unsigned* Ri1p = (unsigned*)take(16 * 256 * 8 * 4);
    unsigned* Ri2p = (unsigned*)take(16 * 256 * 8 * 4);
    unsigned* Ro1p = (unsigned*)take(16 * GFEAT_PAD * 8 * 4);
    unsigned* Ro2p = (unsigned*)take(16 * GFEAT_PAD * 8 * 4);
    float*    gacc = (float*)take((size_t)NG * GFEAT * 4);
    unsigned short* hC = (unsigned short*)hA;   // reuse: mid readout activations
    if (off > ws_size) return;                  // workspace too small; bail

    // ---- weight prep ----
    pack_weight_frag<<<(16 * 256 * 8) / 256, 256, 0, stream>>>(W1, W1p);
    pack_weight_frag<<<(16 * 256 * 8) / 256, 256, 0, stream>>>(Wr1, Wr1p);
    pack_weight_frag<<<(16 * 256 * 8) / 256, 256, 0, stream>>>(W2, W2p);
    pack_weight_frag<<<(16 * 256 * 8) / 256, 256, 0, stream>>>(Wr2, Wr2p);
    pack_weight_frag<<<(16 * 256 * 8) / 256, 256, 0, stream>>>(Ri1, Ri1p);
    pack_weight_frag<<<(16 * 256 * 8) / 256, 256, 0, stream>>>(Ri2, Ri2p);
    pack_ro_frag<<<(16 * GFEAT_PAD * 8 + 255) / 256, 256, 0, stream>>>(Ro1, Ro1p);
    pack_ro_frag<<<(16 * GFEAT_PAD * 8 + 255) / 256, 256, 0, stream>>>(Ro2, Ro2p);
    hipMemsetAsync(gacc, 0, (size_t)NG * GFEAT * 4, stream);

    const int featN = N_NODES * 128;
    const dim3 gemmGrid(MBLOCKS32);
    const dim3 roGrid(MTILES16, 13);

    // ---- graph 1 ----
    pack_feats<<<(featN + 255) / 256, 256, 0, stream>>>(nf1, hA, featN);
    gemm256<0><<<gemmGrid, 128, 0, stream>>>(hA, W1p, nullptr, nullptr, nullptr, F1, nullptr);
    hipMemsetAsync(F2, 0, (size_t)N_NODES * 256 * 4, stream);
    edge_agg<<<(N_EDGES * 64) / 256, 256, 0, stream>>>((const float4*)F1, src1, dst1, F2);
    gemm256<1><<<gemmGrid, 128, 0, stream>>>(hA, Wr1p, F2, b1, br1, nullptr, hB);
    gemm256<2><<<gemmGrid, 128, 0, stream>>>((const unsigned*)hB, Ri1p, nullptr, rbi1, nullptr, nullptr, hC);
    gemm_readout<<<roGrid, 32, 0, stream>>>((const unsigned*)hC, Ro1p, rbo1, gid1, gacc);

    // ---- graph 2 ----
    pack_feats<<<(featN + 255) / 256, 256, 0, stream>>>(nf2, hA, featN);
    gemm256<0><<<gemmGrid, 128, 0, stream>>>(hA, W2p, nullptr, nullptr, nullptr, F1, nullptr);
    hipMemsetAsync(F2, 0, (size_t)N_NODES * 256 * 4, stream);
    edge_agg<<<(N_EDGES * 64) / 256, 256, 0, stream>>>((const float4*)F1, src2, dst2, F2);
    gemm256<1><<<gemmGrid, 128, 0, stream>>>(hA, Wr2p, F2, b2, br2, nullptr, hB);
    gemm256<2><<<gemmGrid, 128, 0, stream>>>((const unsigned*)hB, Ri2p, nullptr, rbi2, nullptr, nullptr, hC);
    gemm_readout<<<roGrid, 32, 0, stream>>>((const unsigned*)hC, Ro2p, rbo2, gid2, gacc);

    // ---- predictor ----
    predict_k<<<2, 256, 0, stream>>>(gacc, Wp, bp, out);
}
